// CrossAttention2D_37228776521788
// MI455X (gfx1250) — matrix operand order repaired and checked by reading, not compiled, not run on hardware
//
#include <hip/hip_runtime.h>
#include <hip/hip_bf16.h>
#include <math.h>

typedef __attribute__((ext_vector_type(16))) _Float16 v16h;
typedef __attribute__((ext_vector_type(8)))  _Float16 v8h;
typedef __attribute__((ext_vector_type(8)))  float    v8f;

#define B_    2
#define C_    256
#define H_    64
#define W_    64
#define NPOS  4096
#define HEADS 8
#define DHEAD 64
#define HID   512
#define KVC   257
#define HAPEC 67
#define SCALE_Q 0.125f   /* 64^-0.5 */

/* ---------------- WMMA helpers (CDNA5 16x16x32 f16 -> f32) -------------- */

__device__ __forceinline__ v8f wmma_acc(v16h a, v16h b, v8f c) {
  return __builtin_amdgcn_wmma_f32_16x16x32_f16(false, a, false, b, (short)0, c,
                                                false, false);
}

/* A fragment: 16x32 (MxK), A row-major [m][k].
   lane l: m = l&15 ; chunks k = kb..kb+7 and kb+16..kb+23, kb = (l<16)?0:8 */
__device__ __forceinline__ v16h load_a_frag(const _Float16* A, int lda) {
  int l  = threadIdx.x & 31;
  int m  = l & 15;
  int kb = (l < 16) ? 0 : 8;
  const _Float16* p = A + m * lda + kb;
  union { v16h v; v8h h[2]; } u;
  u.h[0] = *(const v8h*)(p);
  u.h[1] = *(const v8h*)(p + 16);
  return u.v;
}

/* B fragment: 32x16 (KxN), stored transposed row-major Bt[n][k].
   lane l: n = l&15 ; chunks k = kb..kb+7 and kb+16..kb+23, kb = (l<16)?0:8 */
__device__ __forceinline__ v16h load_b_frag(const _Float16* Bt, int ldb) {
  int l  = threadIdx.x & 31;
  int n  = l & 15;
  int ko = (l < 16) ? 0 : 8;
  const _Float16* p = Bt + n * ldb + ko;
  union { v16h v; v8h h[2]; } u;
  u.h[0] = *(const v8h*)(p);
  u.h[1] = *(const v8h*)(p + 16);
  return u.v;
}

/* ------- Register-blocked batched f16 GEMM: D = A(MxK) * Bt(NxK)^T ------- */
/* 64x64 output per wave (4x4 tiles of 16x16). A-frags loaded once per
   k-step; B-frags streamed one at a time so only ~180 VGPRs are live
   (no scratch spills). launch_bounds(...,1) relaxes the occupancy target
   so the allocator gets the full budget. */
__global__ void __launch_bounds__(128, 1)
gemm_f16_blocked(const _Float16* __restrict__ A,
                 const _Float16* __restrict__ Bt,
                 float* __restrict__ D,
                 int M, int N, int K,
                 long strideB, long strideD) {
  int bz = blockIdx.z;
  const _Float16* Bb = Bt + (long)bz * strideB;
  float* Db = D + (long)bz * strideD;

  int wave = blockIdx.x * (blockDim.x >> 5) + (threadIdx.x >> 5);
  int nbn  = N >> 6;                       /* 64-wide N blocks */
  int mb   = wave / nbn;
  int nb   = wave - mb * nbn;
  if (mb >= (M >> 6)) return;

  const _Float16* Abase = A  + (long)(mb * 64) * K;
  const _Float16* Bbase = Bb + (long)(nb * 64) * K;

  v8f acc[4][4] = {};
  for (int k0 = 0; k0 < K; k0 += 32) {
    v16h af[4];
#pragma unroll
    for (int i = 0; i < 4; ++i)
      af[i] = load_a_frag(Abase + (long)(i * 16) * K + k0, K);
#pragma unroll
    for (int j = 0; j < 4; ++j) {
      v16h bf = load_b_frag(Bbase + (long)(j * 16) * K + k0, K);
      acc[0][j] = wmma_acc(af[0], bf, acc[0][j]);
      acc[1][j] = wmma_acc(af[1], bf, acc[1][j]);
      acc[2][j] = wmma_acc(af[2], bf, acc[2][j]);
      acc[3][j] = wmma_acc(af[3], bf, acc[3][j]);
    }
  }

  int l  = threadIdx.x & 31;
  int n  = l & 15;
  int mo = (l < 16) ? 0 : 8;
#pragma unroll
  for (int i = 0; i < 4; ++i)
#pragma unroll
    for (int j = 0; j < 4; ++j) {
      float* out = Db + (long)(mb * 64 + i * 16 + mo) * N + nb * 64 + j * 16 + n;
#pragma unroll
      for (int v = 0; v < 8; ++v) out[(long)v * N] = acc[i][j][v];
    }
}

/* ---------------- Flash attention: one wave per 32-row Q tile ------------ */
/* Q16: [b][h][pos][d], K16: [b][h][pos][d], Vt16: [b][h][d][pos],
   O16 out: [b][pos][hid] f16 (ready as Bt operand for out-projection).
   2 q-tiles per wave -> each K-frag / V-frag feeds 2 WMMAs; fragments are
   streamed one at a time to keep peak VGPR pressure under the budget. */
__global__ void __launch_bounds__(32, 1)
attn_kernel(const _Float16* __restrict__ Q16,
            const _Float16* __restrict__ K16,
            const _Float16* __restrict__ Vt16,
            _Float16* __restrict__ O16) {
  __shared__ __align__(16) _Float16 lds_p[2 * 16 * 32];

  int blk   = blockIdx.x;
  int itile = blk & 127;          /* 4096/32 Q tiles of 32 rows */
  int bh    = blk >> 7;           /* 0..15 */
  int b     = bh >> 3;
  int h     = bh & 7;

  const _Float16* Qb = Q16 + (long)bh * NPOS * DHEAD + itile * 32 * DHEAD;
  const _Float16* Kb = K16 + (long)bh * NPOS * DHEAD;
  const _Float16* Vb = Vt16 + (long)bh * DHEAD * NPOS;

  int l    = threadIdx.x & 31;
  int half = (l < 16) ? 0 : 1;
  int n    = l & 15;

  v16h aq[2][2];
#pragma unroll
  for (int t = 0; t < 2; ++t) {
    aq[t][0] = load_a_frag(Qb + t * 16 * DHEAD + 0,  DHEAD);
    aq[t][1] = load_a_frag(Qb + t * 16 * DHEAD + 32, DHEAD);
  }

  v8f oacc[2][4] = {};
  float rowmax[2][8], rowsum[2][8];
#pragma unroll
  for (int t = 0; t < 2; ++t)
#pragma unroll
    for (int v = 0; v < 8; ++v) { rowmax[t][v] = -3.0e38f; rowsum[t][v] = 0.0f; }

  for (int j0 = 0; j0 < NPOS; j0 += 32) {
    /* S tiles: stream each K fragment straight into its two WMMAs */
    v8f s[2][2] = {};
    {
      v16h bk = load_b_frag(Kb + (long)(j0 + 0) * DHEAD + 0, DHEAD);
      s[0][0] = wmma_acc(aq[0][0], bk, s[0][0]);
      s[1][0] = wmma_acc(aq[1][0], bk, s[1][0]);
    }
    {
      v16h bk = load_b_frag(Kb + (long)(j0 + 0) * DHEAD + 32, DHEAD);
      s[0][0] = wmma_acc(aq[0][1], bk, s[0][0]);
      s[1][0] = wmma_acc(aq[1][1], bk, s[1][0]);
    }
    {
      v16h bk = load_b_frag(Kb + (long)(j0 + 16) * DHEAD + 0, DHEAD);
      s[0][1] = wmma_acc(aq[0][0], bk, s[0][1]);
      s[1][1] = wmma_acc(aq[1][0], bk, s[1][1]);
    }
    {
      v16h bk = load_b_frag(Kb + (long)(j0 + 16) * DHEAD + 32, DHEAD);
      s[0][1] = wmma_acc(aq[0][1], bk, s[0][1]);
      s[1][1] = wmma_acc(aq[1][1], bk, s[1][1]);
    }

    /* online softmax; row r = v + half*8 spans 16 lanes of one VGPR */
#pragma unroll
    for (int t = 0; t < 2; ++t) {
#pragma unroll
      for (int v = 0; v < 8; ++v) {
        float a  = s[t][0][v];
        float bb = s[t][1][v];
        float mx = fmaxf(a, bb);
#pragma unroll
        for (int off = 1; off < 16; off <<= 1)
          mx = fmaxf(mx, __shfl_xor(mx, off, 32));
        float nm   = fmaxf(rowmax[t][v], mx);
        float corr = __expf(rowmax[t][v] - nm);
        float p0   = __expf(a  - nm);
        float p1   = __expf(bb - nm);
        float ps   = p0 + p1;
#pragma unroll
        for (int off = 1; off < 16; off <<= 1)
          ps += __shfl_xor(ps, off, 32);
        rowsum[t][v] = rowsum[t][v] * corr + ps;
        rowmax[t][v] = nm;
        oacc[t][0][v] *= corr; oacc[t][1][v] *= corr;
        oacc[t][2][v] *= corr; oacc[t][3][v] *= corr;

        int m = v + half * 8;
        lds_p[t * 512 + m * 32 + n]      = (_Float16)p0;
        lds_p[t * 512 + m * 32 + 16 + n] = (_Float16)p1;
      }
    }
    __syncthreads();

    /* P (16x32) as A operands from LDS; stream each V fragment into its
       two WMMAs */
    v16h ap0 = load_a_frag(lds_p + 0,   32);
    v16h ap1 = load_a_frag(lds_p + 512, 32);
#pragma unroll
    for (int nt = 0; nt < 4; ++nt) {
      v16h bv = load_b_frag(Vb + (long)(nt * 16) * NPOS + j0, NPOS);
      oacc[0][nt] = wmma_acc(ap0, bv, oacc[0][nt]);
      oacc[1][nt] = wmma_acc(ap1, bv, oacc[1][nt]);
    }
    __syncthreads();
  }

#pragma unroll
  for (int t = 0; t < 2; ++t) {
    _Float16* Ob = O16 + ((long)b * NPOS + itile * 32 + t * 16) * HID + h * DHEAD + n;
#pragma unroll
    for (int v = 0; v < 8; ++v) {
      float inv = 1.0f / rowsum[t][v];
      _Float16* p = Ob + (long)(v + half * 8) * HID;
      p[0]  = (_Float16)(oacc[t][0][v] * inv);
      p[16] = (_Float16)(oacc[t][1][v] * inv);
      p[32] = (_Float16)(oacc[t][2][v] * inv);
      p[48] = (_Float16)(oacc[t][3][v] * inv);
    }
  }
}

/* ---------------- Elementwise / prep kernels ----------------------------- */

__global__ void minmax_kernel(const float* __restrict__ dsm, float* __restrict__ stat) {
  __shared__ float smin[256], smax[256];
  int b = blockIdx.x;
  const float* p = dsm + b * NPOS;
  float mn = 3.0e38f, mx = -3.0e38f;
  for (int i = threadIdx.x; i < NPOS; i += 256) {
    float v = p[i];
    mn = fminf(mn, v);
    mx = fmaxf(mx, v);
  }
  smin[threadIdx.x] = mn; smax[threadIdx.x] = mx;
  __syncthreads();
  for (int s = 128; s > 0; s >>= 1) {
    if (threadIdx.x < s) {
      smin[threadIdx.x] = fminf(smin[threadIdx.x], smin[threadIdx.x + s]);
      smax[threadIdx.x] = fmaxf(smax[threadIdx.x], smax[threadIdx.x + s]);
    }
    __syncthreads();
  }
  if (threadIdx.x == 0) { stat[b * 2] = smin[0]; stat[b * 2 + 1] = smax[0]; }
}

__global__ void dnorm_kernel(const float* __restrict__ dsm,
                             const float* __restrict__ stat,
                             float* __restrict__ f_d) {
  int tid = blockIdx.x * blockDim.x + threadIdx.x;
  if (tid >= B_ * NPOS) return;
  int b = tid / NPOS;
  float mn = stat[b * 2], mx = stat[b * 2 + 1];
  f_d[tid] = (dsm[tid] - mn) / (mx - mn + 1e-6f);
}

__device__ __forceinline__ float padget(const float* p, int y, int x) {
  if (y < 0 || y >= H_ || x < 0 || x >= W_) return 0.0f;
  return p[y * W_ + x];
}

__global__ void sobel_kernel(const float* __restrict__ f_d,
                             float* __restrict__ gx, float* __restrict__ gy,
                             float* __restrict__ mag) {
  int tid = blockIdx.x * blockDim.x + threadIdx.x;
  if (tid >= B_ * NPOS) return;
  int b = tid / NPOS, pos = tid - b * NPOS;
  int y = pos >> 6, x = pos & 63;
  const float* p = f_d + b * NPOS;
  float vx = padget(p, y - 1, x - 1) - padget(p, y - 1, x + 1)
           + 2.0f * padget(p, y, x - 1) - 2.0f * padget(p, y, x + 1)
           + padget(p, y + 1, x - 1) - padget(p, y + 1, x + 1);
  float vy = padget(p, y - 1, x - 1) + 2.0f * padget(p, y - 1, x) + padget(p, y - 1, x + 1)
           - padget(p, y + 1, x - 1) - 2.0f * padget(p, y + 1, x) - padget(p, y + 1, x + 1);
  gx[tid] = vx; gy[tid] = vy;
  float rx = fmaxf(vx, 0.0f), ry = fmaxf(vy, 0.0f);
  mag[tid] = __fsqrt_rn(rx * rx + ry * ry + 1e-12f);
}

__global__ void alpha_kernel(const float* __restrict__ mag,
                             const float* __restrict__ w_alpha,
                             const float* __restrict__ b_alpha,
                             float* __restrict__ alpha) {
  int tid = blockIdx.x * blockDim.x + threadIdx.x;
  if (tid >= B_ * NPOS) return;
  int b = tid / NPOS, pos = tid - b * NPOS;
  int y = pos >> 6, x = pos & 63;
  const float* p = mag + b * NPOS;
  float acc = b_alpha[0];
#pragma unroll
  for (int i = 0; i < 3; ++i)
#pragma unroll
    for (int j = 0; j < 3; ++j)
      acc += w_alpha[i * 3 + j] * padget(p, y + i - 1, x + j - 1);
  alpha[tid] = 1.0f / (1.0f + __expf(-acc));
}

/* hape_in[b][c][pos]: c<64 -> sincos PE, 64 -> d, 65 -> gx, 66 -> gy */
__global__ void hapein_kernel(const float* __restrict__ f_d,
                              const float* __restrict__ gx,
                              const float* __restrict__ gy,
                              float* __restrict__ hin) {
  int tid = blockIdx.x * blockDim.x + threadIdx.x;
  if (tid >= B_ * HAPEC * NPOS) return;
  int b = tid / (HAPEC * NPOS);
  int r = tid - b * (HAPEC * NPOS);
  int c = r / NPOS, pos = r - c * NPOS;
  int y = pos >> 6, x = pos & 63;
  float val;
  if (c < 64) {
    int q = c >> 4, k = c & 15;
    float omega = __powf(10000.0f, -(float)k / 16.0f);
    float xl = -1.0f + 2.0f * (float)x / 63.0f;
    float yl = -1.0f + 2.0f * (float)y / 63.0f;
    float t = ((q < 2) ? xl : yl) * omega;
    val = ((q & 1) == 0) ? __sinf(t) : __cosf(t);
  } else if (c == 64) val = f_d[b * NPOS + pos];
  else if (c == 65)   val = gx[b * NPOS + pos];
  else                val = gy[b * NPOS + pos];
  hin[tid] = val;
}

/* x[b][c][pos] f32 -> xT16[b][pos][c] f16 */
__global__ void cvt_xT_kernel(const float* __restrict__ x, _Float16* __restrict__ xT) {
  int tid = blockIdx.x * blockDim.x + threadIdx.x;
  if (tid >= B_ * NPOS * C_) return;
  int b = tid / (NPOS * C_);
  int r = tid - b * (NPOS * C_);
  int pos = r / C_, c = r - pos * C_;
  xT[tid] = (_Float16)x[((long)b * C_ + c) * NPOS + pos];
}

/* generic f32 row-strided -> f16 dense (rows x cols) */
__global__ void cvt_rows_kernel(const float* __restrict__ src, _Float16* __restrict__ dst,
                                int rows, int cols, int srcStride) {
  int tid = blockIdx.x * blockDim.x + threadIdx.x;
  if (tid >= rows * cols) return;
  int rr = tid / cols, cc = tid - rr * cols;
  dst[tid] = (_Float16)src[(long)rr * srcStride + cc];
}

/* q_f32[b][o][pos] -> Q16[b][h][pos][d], scaled */
__global__ void pack_q_kernel(const float* __restrict__ q, _Float16* __restrict__ Q16) {
  int tid = blockIdx.x * blockDim.x + threadIdx.x;
  if (tid >= B_ * HID * NPOS) return;
  int b = tid / (HID * NPOS);
  int r = tid - b * (HID * NPOS);
  int o = r / NPOS, pos = r - o * NPOS;
  int h = o >> 6, d = o & 63;
  Q16[(((long)(b * HEADS + h) * NPOS) + pos) * DHEAD + d] = (_Float16)(q[tid] * SCALE_Q);
}

/* K = kv[:512] + wkv[:,256]*d + hape(w_hape @ hape_in + b_hape) -> K16[b][h][pos][d] */
__global__ void pack_k_kernel(const float* __restrict__ kv,
                              const float* __restrict__ wkv,
                              const float* __restrict__ f_d,
                              const float* __restrict__ hin,
                              const float* __restrict__ w_hape,
                              const float* __restrict__ b_hape,
                              _Float16* __restrict__ K16) {
  int tid = blockIdx.x * blockDim.x + threadIdx.x;
  if (tid >= B_ * HID * NPOS) return;
  int b = tid / (HID * NPOS);
  int r = tid - b * (HID * NPOS);
  int o = r / NPOS, pos = r - o * NPOS;
  float val = kv[((long)b * 2 * HID + o) * NPOS + pos]
            + wkv[(long)o * KVC + (KVC - 1)] * f_d[b * NPOS + pos]
            + b_hape[o];
  const float* wrow = w_hape + (long)o * HAPEC;
  const float* hb   = hin + (long)b * HAPEC * NPOS + pos;
#pragma unroll 1
  for (int c = 0; c < HAPEC; ++c)
    val += wrow[c] * hb[(long)c * NPOS];
  int h = o >> 6, d = o & 63;
  K16[(((long)(b * HEADS + h) * NPOS) + pos) * DHEAD + d] = (_Float16)val;
}

/* V = kv[512:] + wkv[512:,256]*d -> Vt16[b][h][d][pos] (layout == [b][o][pos]) */
__global__ void pack_v_kernel(const float* __restrict__ kv,
                              const float* __restrict__ wkv,
                              const float* __restrict__ f_d,
                              _Float16* __restrict__ Vt16) {
  int tid = blockIdx.x * blockDim.x + threadIdx.x;
  if (tid >= B_ * HID * NPOS) return;
  int b = tid / (HID * NPOS);
  int r = tid - b * (HID * NPOS);
  int o = r / NPOS, pos = r - o * NPOS;
  int o2 = HID + o;
  float val = kv[((long)b * 2 * HID + o2) * NPOS + pos]
            + wkv[(long)o2 * KVC + (KVC - 1)] * f_d[b * NPOS + pos];
  Vt16[tid] = (_Float16)val;
}

/* final = x + (outproj + bout) * alpha */
__global__ void epilogue_kernel(const float* __restrict__ x,
                                const float* __restrict__ proj,
                                const float* __restrict__ bout,
                                const float* __restrict__ alpha,
                                float* __restrict__ out) {
  int tid = blockIdx.x * blockDim.x + threadIdx.x;
  if (tid >= B_ * C_ * NPOS) return;
  int b = tid / (C_ * NPOS);
  int r = tid - b * (C_ * NPOS);
  int c = r / NPOS, pos = r - c * NPOS;
  out[tid] = x[tid] + (proj[tid] + bout[c]) * alpha[b * NPOS + pos];
}

/* ---------------- Host side --------------------------------------------- */

extern "C" void kernel_launch(void* const* d_in, const int* in_sizes, int n_in,
                              void* d_out, int out_size, void* d_ws, size_t ws_size,
                              hipStream_t stream) {
  (void)in_sizes; (void)n_in; (void)out_size; (void)ws_size;
  const float* x       = (const float*)d_in[0];
  const float* dsm     = (const float*)d_in[1];
  const float* wq      = (const float*)d_in[2];
  const float* wkv     = (const float*)d_in[3];
  const float* wout    = (const float*)d_in[4];
  const float* bout    = (const float*)d_in[5];
  const float* w_hape  = (const float*)d_in[6];
  const float* b_hape  = (const float*)d_in[7];
  const float* w_alpha = (const float*)d_in[8];
  const float* b_alpha = (const float*)d_in[9];
  float* out = (float*)d_out;
  char* ws = (char*)d_ws;

  size_t off = 0;
  auto alloc = [&](size_t bytes) { size_t r = off; off += (bytes + 255) & ~(size_t)255; return r; };

  float*    f_stat  = (float*)   (ws + alloc(64));
  float*    f_d     = (float*)   (ws + alloc((size_t)B_ * NPOS * 4));
  float*    f_gx    = (float*)   (ws + alloc((size_t)B_ * NPOS * 4));
  float*    f_gy    = (float*)   (ws + alloc((size_t)B_ * NPOS * 4));
  float*    f_mag   = (float*)   (ws + alloc((size_t)B_ * NPOS * 4));
  float*    f_alpha = (float*)   (ws + alloc((size_t)B_ * NPOS * 4));
  float*    hin     = (float*)   (ws + alloc((size_t)B_ * HAPEC * NPOS * 4));
  _Float16* xT16    = (_Float16*)(ws + alloc((size_t)B_ * NPOS * C_ * 2));
  _Float16* wq16    = (_Float16*)(ws + alloc((size_t)HID * C_ * 2));
  _Float16* wkv16   = (_Float16*)(ws + alloc((size_t)2 * HID * C_ * 2));
  _Float16* wout16  = (_Float16*)(ws + alloc((size_t)C_ * HID * 2));
  float*    q_f32   = (float*)   (ws + alloc((size_t)B_ * HID * NPOS * 4));
  float*    kv_f32  = (float*)   (ws + alloc((size_t)B_ * 2 * HID * NPOS * 4));
  _Float16* Q16     = (_Float16*)(ws + alloc((size_t)B_ * HID * NPOS * 2));
  _Float16* K16     = (_Float16*)(ws + alloc((size_t)B_ * HID * NPOS * 2));
  _Float16* Vt16    = (_Float16*)(ws + alloc((size_t)B_ * HID * NPOS * 2));
  _Float16* O16     = (_Float16*)(ws + alloc((size_t)B_ * NPOS * HID * 2));
  float*    proj    = (float*)   (ws + alloc((size_t)B_ * C_ * NPOS * 4));

  auto nb = [](int total) { return (total + 255) / 256; };

  /* prep */
  minmax_kernel<<<B_, 256, 0, stream>>>(dsm, f_stat);
  dnorm_kernel<<<nb(B_ * NPOS), 256, 0, stream>>>(dsm, f_stat, f_d);
  sobel_kernel<<<nb(B_ * NPOS), 256, 0, stream>>>(f_d, f_gx, f_gy, f_mag);
  alpha_kernel<<<nb(B_ * NPOS), 256, 0, stream>>>(f_mag, w_alpha, b_alpha, f_alpha);
  hapein_kernel<<<nb(B_ * HAPEC * NPOS), 256, 0, stream>>>(f_d, f_gx, f_gy, hin);

  /* conversions to f16 */
  cvt_xT_kernel<<<nb(B_ * NPOS * C_), 256, 0, stream>>>(x, xT16);
  cvt_rows_kernel<<<nb(HID * C_), 256, 0, stream>>>(wq, wq16, HID, C_, C_);
  cvt_rows_kernel<<<nb(2 * HID * C_), 256, 0, stream>>>(wkv, wkv16, 2 * HID, C_, KVC);
  cvt_rows_kernel<<<nb(C_ * HID), 256, 0, stream>>>(wout, wout16, C_, HID, HID);

  /* projection GEMMs (WMMA, 64x64 register-blocked waves) */
  {
    dim3 blk(128, 1, 1);
    int waves_q = (HID / 64) * (NPOS / 64);
    gemm_f16_blocked<<<dim3(waves_q / 4, 1, B_), blk, 0, stream>>>(
        wq16, xT16, q_f32, HID, NPOS, C_, (long)NPOS * C_, (long)HID * NPOS);
    int waves_kv = (2 * HID / 64) * (NPOS / 64);
    gemm_f16_blocked<<<dim3(waves_kv / 4, 1, B_), blk, 0, stream>>>(
        wkv16, xT16, kv_f32, 2 * HID, NPOS, C_, (long)NPOS * C_, (long)2 * HID * NPOS);
  }

  /* pack q/k/v into attention layouts */
  pack_q_kernel<<<nb(B_ * HID * NPOS), 256, 0, stream>>>(q_f32, Q16);
  pack_k_kernel<<<nb(B_ * HID * NPOS), 256, 0, stream>>>(kv_f32, wkv, f_d, hin,
                                                         w_hape, b_hape, K16);
  pack_v_kernel<<<nb(B_ * HID * NPOS), 256, 0, stream>>>(kv_f32, wkv, f_d, Vt16);

  /* flash attention: one wave per 32-row Q tile, 16 (b,h) pairs */
  attn_kernel<<<B_ * HEADS * (NPOS / 32), 32, 0, stream>>>(Q16, K16, Vt16, O16);

  /* out projection (WMMA) */
  {
    dim3 blk(128, 1, 1);
    int waves_o = (C_ / 64) * (NPOS / 64);
    gemm_f16_blocked<<<dim3(waves_o / 4, 1, B_), blk, 0, stream>>>(
        wout16, O16, proj, C_, NPOS, HID, (long)NPOS * HID, (long)C_ * NPOS);
  }

  /* epilogue */
  epilogue_kernel<<<nb(B_ * C_ * NPOS), 256, 0, stream>>>(x, proj, bout, f_alpha, out);
}
